// User_simmulation_Model_50611894616443
// MI455X (gfx1250) — matrix operand order repaired and checked by reading, not compiled, not run on hardware
//
#include <hip/hip_runtime.h>

// ---------------------------------------------------------------------------
// Types for CDNA5 WMMA (wave32, 16x16x32 bf16 -> f32)
// ---------------------------------------------------------------------------
typedef __attribute__((ext_vector_type(16))) __bf16 v16bf;
typedef __attribute__((ext_vector_type(8)))  __bf16 v8bf;
typedef __attribute__((ext_vector_type(8)))  float  v8f;

#define WMMA_BF16(a, b, c) \
  __builtin_amdgcn_wmma_f32_16x16x32_bf16(false, (a), false, (b), (short)0, (c), false, false)

#define H_USER   0.1f
#define H_GLOBAL 0.05f
#define EPS_F    1e-6f

__device__ __forceinline__ float silu_f(float x) { return x / (1.0f + __expf(-x)); }
__device__ __forceinline__ float sigmoid_f(float x) { return 1.0f / (1.0f + __expf(-x)); }

// ---------------------------------------------------------------------------
// LDS layout for the main kernel (dynamic shared memory)
// ---------------------------------------------------------------------------
constexpr int MT = 32;     // rows (batch tile) per workgroup
constexpr int SP = 264;    // state pitch (elements, 256 + 8 pad)
constexpr int HP = 1032;   // hidden pitch (elements, 1024 + 8 pad)

constexpr size_t OFF_SSTATE = 0;
constexpr size_t OFF_SACT   = OFF_SSTATE + (size_t)MT * SP * 4;  // f32 state
constexpr size_t OFF_HACT   = OFF_SACT   + (size_t)MT * SP * 2;  // bf16 state
constexpr size_t OFF_SB0    = OFF_HACT   + (size_t)MT * HP * 2;  // bf16 hidden
constexpr size_t OFF_UB0    = OFF_SB0  + 4096;
constexpr size_t OFF_UB1    = OFF_UB0  + 4096;
constexpr size_t OFF_UW2    = OFF_UB1  + 4096;
constexpr size_t OFF_SB1    = OFF_UW2  + 4096;
constexpr size_t OFF_STEP   = OFF_SB1  + 1024;
constexpr size_t OFF_ACC    = OFF_STEP + 128;
constexpr size_t SMEM_BYTES = OFF_ACC  + 128;   // ~134.4 KB -> 2 WGs / WGP

// ---------------------------------------------------------------------------
// Fragment loaders (16-bit A and B, contiguous K order k = 16*(lane>>4) + s)
// ---------------------------------------------------------------------------
// A (16x32 bf16): lane L holds row M=L&15; K = (L>>4)*16 .. +15.
__device__ __forceinline__ v16bf load_a_frag(const __bf16* act, int pitch,
                                             int rowBlock, int kb, int lane) {
  int m  = rowBlock * 16 + (lane & 15);
  int k0 = kb * 32 + ((lane >> 4) << 4);
  const __bf16* p = act + m * pitch + k0;
  v8bf lo = *(const v8bf*)(p);
  v8bf hi = *(const v8bf*)(p + 8);
  return __builtin_shufflevector(lo, hi, 0, 1, 2, 3, 4, 5, 6, 7,
                                 8, 9, 10, 11, 12, 13, 14, 15);
}

// B fragments are pre-swizzled in global memory: frag*512 + lane*16 elements.
__device__ __forceinline__ v16bf load_b_frag(const __bf16* wf, int frag, int lane) {
  return *(const v16bf*)(wf + (size_t)frag * 512 + lane * 16);
}

// One wave computes 4 adjacent 16x16 output blocks per K sweep.
// B fragments are double-buffered: next kb's loads are issued before the
// current WMMAs so in-order VMEM completion lets the waits stay shallow
// (s_wait_loadcnt <= in-flight prefetches instead of 0).
template <int KB>
__device__ __forceinline__ void gemm_tile4(const __bf16* act, int pitch, int waveM,
                                           const __bf16* wf, int nb0, int lane,
                                           v8f acc4[4]) {
  v16bf b[4];
#pragma unroll
  for (int j = 0; j < 4; ++j) b[j] = load_b_frag(wf, (nb0 + j) * KB, lane);

  for (int kb = 0; kb < KB - 1; ++kb) {
    v16bf a = load_a_frag(act, pitch, waveM, kb, lane);
    v16bf bn[4];
#pragma unroll
    for (int j = 0; j < 4; ++j)
      bn[j] = load_b_frag(wf, (nb0 + j) * KB + kb + 1, lane);
#pragma unroll
    for (int j = 0; j < 4; ++j) acc4[j] = WMMA_BF16(a, b[j], acc4[j]);
#pragma unroll
    for (int j = 0; j < 4; ++j) b[j] = bn[j];
  }
  v16bf a = load_a_frag(act, pitch, waveM, KB - 1, lane);
#pragma unroll
  for (int j = 0; j < 4; ++j) acc4[j] = WMMA_BF16(a, b[j], acc4[j]);
}

// ---------------------------------------------------------------------------
// Prepass: convert f32 weights -> bf16 WMMA B fragments.
// Fragment (kb, nb) covers K in [kb*32, +32), N in [nb*16, +16).
// Lane l supplies column n = nb*16 + (l&15), K = kb*32 + (l>>4)*16 .. +15.
// ---------------------------------------------------------------------------
__global__ void swizzle_weight_kernel(const float* __restrict__ src,
                                      __bf16* __restrict__ dst, int K, int N) {
  const int frag = blockIdx.x;
  const int KB = K >> 5;
  const int nb = frag / KB;
  const int kb = frag - nb * KB;
  const int lane = threadIdx.x;
  const int n = nb * 16 + (lane & 15);
  const int kbase = kb * 32 + ((lane >> 4) << 4);
  __bf16* out = dst + (size_t)frag * 512 + lane * 16;
#pragma unroll
  for (int j = 0; j < 16; ++j) {
    out[j] = (__bf16)src[(size_t)(kbase + j) * N + n];
  }
}

// ---------------------------------------------------------------------------
// Global intensity: tiny scalar MLP 1 -> 1024 -> 1024 -> 1, sigmoid-scaled.
// ---------------------------------------------------------------------------
__global__ void global_intensity_kernel(const float* __restrict__ timep,
                                        const float* __restrict__ gW0,
                                        const float* __restrict__ gb0,
                                        const float* __restrict__ gW1,
                                        const float* __restrict__ gb1,
                                        const float* __restrict__ gW2,
                                        const float* __restrict__ gb2,
                                        const int* __restrict__ nInt,
                                        float* __restrict__ gI) {
  __shared__ float h0[1024];
  __shared__ float h1[1024];
  const int t = threadIdx.x;
  const float tv = timep[0];
  h0[t] = silu_f(tv * gW0[t] + gb0[t]);
  __syncthreads();
  float acc = gb1[t];
  for (int k = 0; k < 1024; ++k) acc += h0[k] * gW1[k * 1024 + t];
  h1[t] = silu_f(acc) * gW2[t];
  __syncthreads();
  for (int s = 512; s > 0; s >>= 1) {
    if (t < s) h1[t] += h1[t + s];
    __syncthreads();
  }
  if (t == 0) {
    float g = h1[0] + gb2[0];
    float scale = 2.0f * (float)nInt[0] * H_GLOBAL;
    gI[0] = sigmoid_f(g * scale);
  }
}

// ---------------------------------------------------------------------------
// Main kernel: each workgroup owns 32 batch rows and runs all Euler steps.
// 256 threads = 8 waves arranged as 2 (row blocks) x 4 (column strips).
// ---------------------------------------------------------------------------
__global__ void __launch_bounds__(256)
user_ode_kernel(const float* __restrict__ state,
                const float* __restrict__ sb0,
                const float* __restrict__ sb1,
                const float* __restrict__ ub0,
                const float* __restrict__ ub1,
                const float* __restrict__ uW2,
                const float* __restrict__ ub2,
                const __bf16* __restrict__ sw0f,
                const __bf16* __restrict__ sw1f,
                const __bf16* __restrict__ uw0f,
                const __bf16* __restrict__ uw1f,
                const float* __restrict__ gI,
                const int* __restrict__ nIntervals,
                float* __restrict__ out) {
  extern __shared__ unsigned char smem[];
  float*  sState  = (float*)(smem + OFF_SSTATE);
  __bf16* sAct    = (__bf16*)(smem + OFF_SACT);
  __bf16* hAct    = (__bf16*)(smem + OFF_HACT);
  float*  sb0L    = (float*)(smem + OFF_SB0);
  float*  ub0L    = (float*)(smem + OFF_UB0);
  float*  ub1L    = (float*)(smem + OFF_UB1);
  float*  uw2L    = (float*)(smem + OFF_UW2);
  float*  sb1L    = (float*)(smem + OFF_SB1);
  float*  stepAcc = (float*)(smem + OFF_STEP);
  float*  accL    = (float*)(smem + OFF_ACC);

  const int tid   = threadIdx.x;
  const int lane  = tid & 31;
  const int wave  = tid >> 5;
  const int waveM = wave & 1;       // which 16-row block
  const int waveN = wave >> 1;      // which column strip (0..3)
  const int wg    = blockIdx.x;
  const int colLane  = lane & 15;                       // N within 16x16 block
  const int rowBase0 = waveM * 16 + ((lane >> 4) << 3); // M base for C frag

  // Preload biases / uW2 and the state tile.
  for (int i = tid; i < 1024; i += 256) {
    sb0L[i] = sb0[i];
    ub0L[i] = ub0[i];
    ub1L[i] = ub1[i];
    uw2L[i] = uW2[i];
  }
  sb1L[tid] = sb1[tid & 255];
  for (int e = tid; e < MT * 256; e += 256) {
    int m = e >> 8, c = e & 255;
    float v = state[(size_t)(wg * MT + m) * 256 + c];
    sState[m * SP + c] = v;
    sAct[m * SP + c]   = (__bf16)v;
  }
  if (tid < MT) accL[tid] = 0.0f;
  __syncthreads();

  const int   nSteps = nIntervals[0];
  const float ub2v   = ub2[0];
  const v8f   zero8  = {0.f, 0.f, 0.f, 0.f, 0.f, 0.f, 0.f, 0.f};

  for (int step = 0; step < nSteps; ++step) {
    // ---- G1: h0 = silu(s @ sW0 + sb0)   [32,256]x[256,1024] ----
    for (int i = 0; i < 4; ++i) {
      int nb0 = waveN * 16 + i * 4;
      v8f c4[4];
#pragma unroll
      for (int j = 0; j < 4; ++j) c4[j] = zero8;
      gemm_tile4<8>(sAct, SP, waveM, sw0f, nb0, lane, c4);
#pragma unroll
      for (int j = 0; j < 4; ++j) {
        int col = (nb0 + j) * 16 + colLane;
        float b = sb0L[col];
#pragma unroll
        for (int r = 0; r < 8; ++r)
          hAct[(rowBase0 + r) * HP + col] = (__bf16)silu_f(c4[j][r] + b);
      }
    }
    __syncthreads();

    // ---- G2: s += H_USER * (h0 @ sW1 + sb1)   [32,1024]x[1024,256] ----
    {
      int nb0 = waveN * 4;
      v8f c4[4];
#pragma unroll
      for (int j = 0; j < 4; ++j) c4[j] = zero8;
      gemm_tile4<32>(hAct, HP, waveM, sw1f, nb0, lane, c4);
#pragma unroll
      for (int j = 0; j < 4; ++j) {
        int col = (nb0 + j) * 16 + colLane;
        float b = sb1L[col];
#pragma unroll
        for (int r = 0; r < 8; ++r) {
          int idx = (rowBase0 + r) * SP + col;
          float sn = sState[idx] + H_USER * (c4[j][r] + b);
          sState[idx] = sn;
          sAct[idx]   = (__bf16)sn;
        }
      }
    }
    __syncthreads();
    if (tid < MT) stepAcc[tid] = 0.0f;

    // ---- G3: h1 = silu(s' @ uW0 + ub0)   [32,256]x[256,1024] ----
    for (int i = 0; i < 4; ++i) {
      int nb0 = waveN * 16 + i * 4;
      v8f c4[4];
#pragma unroll
      for (int j = 0; j < 4; ++j) c4[j] = zero8;
      gemm_tile4<8>(sAct, SP, waveM, uw0f, nb0, lane, c4);
#pragma unroll
      for (int j = 0; j < 4; ++j) {
        int col = (nb0 + j) * 16 + colLane;
        float b = ub0L[col];
#pragma unroll
        for (int r = 0; r < 8; ++r)
          hAct[(rowBase0 + r) * HP + col] = (__bf16)silu_f(c4[j][r] + b);
      }
    }
    __syncthreads();

    // ---- G4 (+fused G5): partial[M] += silu(h1@uW1 + ub1)[M,n] * uW2[n] ----
    {
      float partial[8];
#pragma unroll
      for (int r = 0; r < 8; ++r) partial[r] = 0.0f;
      for (int i = 0; i < 4; ++i) {
        int nb0 = waveN * 16 + i * 4;
        v8f c4[4];
#pragma unroll
        for (int j = 0; j < 4; ++j) c4[j] = zero8;
        gemm_tile4<32>(hAct, HP, waveM, uw1f, nb0, lane, c4);
#pragma unroll
        for (int j = 0; j < 4; ++j) {
          int col = (nb0 + j) * 16 + colLane;
          float b = ub1L[col];
          float w = uw2L[col];
#pragma unroll
          for (int r = 0; r < 8; ++r)
            partial[r] += silu_f(c4[j][r] + b) * w;
        }
      }
      // Reduce across the 16 lanes sharing each row, then one LDS atomic.
#pragma unroll
      for (int r = 0; r < 8; ++r) {
        float v = partial[r];
        v += __shfl_xor(v, 1, 16);
        v += __shfl_xor(v, 2, 16);
        v += __shfl_xor(v, 4, 16);
        v += __shfl_xor(v, 8, 16);
        if (colLane == 0) atomicAdd(&stepAcc[rowBase0 + r], v);
      }
    }
    __syncthreads();

    // ---- acc += relu(dot + ub2) * H_USER  (ReLU after full reduction) ----
    if (tid < MT) accL[tid] += fmaxf(stepAcc[tid] + ub2v, 0.0f) * H_USER;
    __syncthreads();
  }

  if (tid < MT) {
    float v = sigmoid_f(accL[tid]) + gI[0];
    v = fminf(fmaxf(v, EPS_F), 1.0f - EPS_F);
    out[(size_t)wg * MT + tid] = v;
  }
}

// ---------------------------------------------------------------------------
// Launch: prepass swizzles (deterministic, re-run every call), global MLP,
// then the fused ODE kernel.
// ---------------------------------------------------------------------------
extern "C" void kernel_launch(void* const* d_in, const int* in_sizes, int n_in,
                              void* d_out, int out_size, void* d_ws, size_t ws_size,
                              hipStream_t stream) {
  const float* state = (const float*)d_in[0];
  const float* timep = (const float*)d_in[1];
  const float* uW0 = (const float*)d_in[2];
  const float* ub0 = (const float*)d_in[3];
  const float* uW1 = (const float*)d_in[4];
  const float* ub1 = (const float*)d_in[5];
  const float* uW2 = (const float*)d_in[6];
  const float* ub2 = (const float*)d_in[7];
  const float* sW0 = (const float*)d_in[8];
  const float* sb0 = (const float*)d_in[9];
  const float* sW1 = (const float*)d_in[10];
  const float* sb1 = (const float*)d_in[11];
  const float* gW0 = (const float*)d_in[12];
  const float* gb0 = (const float*)d_in[13];
  const float* gW1 = (const float*)d_in[14];
  const float* gb1 = (const float*)d_in[15];
  const float* gW2 = (const float*)d_in[16];
  const float* gb2 = (const float*)d_in[17];
  const int*   nInt = (const int*)d_in[18];
  float* out = (float*)d_out;

  // Workspace carve (bytes):
  //   uw0f 512 KB | uw1f 2 MB | sw0f 512 KB | sw1f 512 KB | gI 4 B
  unsigned char* ws = (unsigned char*)d_ws;
  __bf16* uw0f = (__bf16*)(ws + 0);
  __bf16* uw1f = (__bf16*)(ws + 524288);
  __bf16* sw0f = (__bf16*)(ws + 524288 + 2097152);
  __bf16* sw1f = (__bf16*)(ws + 524288 + 2097152 + 524288);
  float*  gIp  = (float*)(ws + 524288 + 2097152 + 524288 + 524288);

  // Pre-swizzle weights to bf16 WMMA B-fragment order.
  swizzle_weight_kernel<<<512, 32, 0, stream>>>(uW0, uw0f, 256, 1024);
  swizzle_weight_kernel<<<2048, 32, 0, stream>>>(uW1, uw1f, 1024, 1024);
  swizzle_weight_kernel<<<512, 32, 0, stream>>>(sW0, sw0f, 256, 1024);
  swizzle_weight_kernel<<<512, 32, 0, stream>>>(sW1, sw1f, 1024, 256);

  // Scalar global-intensity MLP.
  global_intensity_kernel<<<1, 1024, 0, stream>>>(timep, gW0, gb0, gW1, gb1,
                                                  gW2, gb2, nInt, gIp);

  // Main fused ODE kernel: one workgroup per 32 batch rows.
  const int B = in_sizes[0] / 256;
  const int nBlocks = B / MT;
  hipFuncSetAttribute((const void*)user_ode_kernel,
                      hipFuncAttributeMaxDynamicSharedMemorySize,
                      (int)SMEM_BYTES);
  user_ode_kernel<<<nBlocks, 256, SMEM_BYTES, stream>>>(
      state, sb0, sb1, ub0, ub1, uW2, ub2,
      sw0f, sw1f, uw0f, uw1f, gIp, nInt, out);
}
